// RingDilatedAttentionSimpleTriton_68959994905217
// MI455X (gfx1250) — compile-verified, matrix operand order checked by reading, not run on hardware
//
#include <hip/hip_runtime.h>

typedef __attribute__((ext_vector_type(16))) __bf16        v16bf;
typedef __attribute__((ext_vector_type(8)))  float         v8f;
typedef __attribute__((ext_vector_type(8)))  unsigned int  v8u;

__device__ __forceinline__ unsigned int pkbf(float a, float b) {
    unsigned short lo = __builtin_bit_cast(unsigned short, (__bf16)a);
    unsigned short hi = __builtin_bit_cast(unsigned short, (__bf16)b);
    return (unsigned int)lo | ((unsigned int)hi << 16);
}

__device__ __forceinline__ v16bf mk_v16bf(uint4 c0, uint4 c1) {
    v8u u;
    u[0] = c0.x; u[1] = c0.y; u[2] = c0.z; u[3] = c0.w;
    u[4] = c1.x; u[5] = c1.y; u[6] = c1.z; u[7] = c1.w;
    return __builtin_bit_cast(v16bf, u);
}

// One attention "branch pass": grid = (16 q-tiles, ns segments, 16 heads),
// 256 threads = 8 waves, each wave owns 16 query rows.
// Block attends 128 query rows over 2048 keys (16 key-tiles of 128), dilated
// gather with stride r from base seg*L + seg%r. bf16 WMMA, fp32 softmax.
__global__ __launch_bounds__(256)
void dilated_attn_pass(const float* __restrict__ q, const float* __restrict__ kmat,
                       const float* __restrict__ vmat, float* __restrict__ out,
                       int L, int r, int accumulate)
{
    constexpr int H = 16, D = 64, TN = 128, NKT = 16;

    // 51.0 KB static LDS: probs buffer aliases the K buffer.
    __shared__ __align__(16) unsigned char smem[34816 + 17408];
    unsigned int (*kbuf)[36]     = reinterpret_cast<unsigned int (*)[36]>(smem);      // 128 keys x 32 uints(+4 pad)
    __bf16      (*pbuf)[16][136] = reinterpret_cast<__bf16 (*)[16][136]>(smem);       // per-wave 16x128(+8) probs
    __bf16      (*vbuf)[136]     = reinterpret_cast<__bf16 (*)[136]>(smem + 34816);   // 64 dims x 128 keys(+8), transposed V

    const int qtile  = blockIdx.x;
    const int seg    = blockIdx.y;
    const int head   = blockIdx.z;
    const int base_s = seg * L + (seg % r);

    const int tid  = threadIdx.x;
    const int wave = tid >> 5;
    const int lane = tid & 31;
    const int nlo  = lane & 15;
    const int koff = (lane < 16) ? 0 : 8;   // 16-bit WMMA K-chunk split per half-wave

    // ---- Q tile: 16 rows per wave, resident in registers in A-operand layout
    const int  jq = qtile * 128 + wave * 16 + nlo;
    const long sq = base_s + (long)jq * r;
    const float* qrow = q + (sq * H + head) * D;
    v16bf qa[2];
#pragma unroll
    for (int h = 0; h < 2; ++h) {
        const float* p0 = qrow + h * 32 + koff;
        float4 f0 = *(const float4*)(p0);
        float4 f1 = *(const float4*)(p0 + 4);
        float4 f2 = *(const float4*)(p0 + 16);
        float4 f3 = *(const float4*)(p0 + 20);
        v8u u;
        u[0] = pkbf(f0.x, f0.y); u[1] = pkbf(f0.z, f0.w);
        u[2] = pkbf(f1.x, f1.y); u[3] = pkbf(f1.z, f1.w);
        u[4] = pkbf(f2.x, f2.y); u[5] = pkbf(f2.z, f2.w);
        u[6] = pkbf(f3.x, f3.y); u[7] = pkbf(f3.z, f3.w);
        qa[h] = __builtin_bit_cast(v16bf, u);
    }

    float mrun[8], lrun[8];
    v8f   oacc[4] = {};
#pragma unroll
    for (int g = 0; g < 8; ++g) { mrun[g] = -3.0e38f; lrun[g] = 0.f; }

    const float SCALE = 0.125f;  // 1/sqrt(64)

    for (int kt = 0; kt < NKT; ++kt) {
        __syncthreads();   // previous iteration's PV done before restaging
        // ---- stage K (bf16, row-major) and V (bf16, transposed) tiles
        {
            const int  key = tid >> 1;
            const int  dh  = (tid & 1) * 32;
            const long sk  = base_s + (long)(kt * TN + key) * r;
            const float* krow = kmat + (sk * H + head) * D + dh;
            const float* vrow = vmat + (sk * H + head) * D + dh;
            if (kt + 1 < NKT) {  // global_prefetch_b8 of next tile
                __builtin_prefetch(krow + (size_t)TN * r * H * D, 0, 0);
                __builtin_prefetch(vrow + (size_t)TN * r * H * D, 0, 0);
            }
            float kv[32] __attribute__((aligned(16)));
            float vv[32] __attribute__((aligned(16)));
#pragma unroll
            for (int i = 0; i < 8; ++i) *(float4*)(kv + 4 * i) = *(const float4*)(krow + 4 * i);
#pragma unroll
            for (int i = 0; i < 8; ++i) *(float4*)(vv + 4 * i) = *(const float4*)(vrow + 4 * i);
            unsigned int* kd = &kbuf[key][dh >> 1];
#pragma unroll
            for (int i = 0; i < 16; ++i) kd[i] = pkbf(kv[2 * i], kv[2 * i + 1]);
#pragma unroll
            for (int i = 0; i < 32; ++i) vbuf[dh + i][key] = (__bf16)vv[i];
        }
        __syncthreads();

        // ---- S = Q K^T : 8 column tiles x 2 K-halves of WMMA per wave
        v8f sc[8];
#pragma unroll
        for (int t = 0; t < 8; ++t) {
            v8f c = {};
#pragma unroll
            for (int h = 0; h < 2; ++h) {
                const unsigned int* kr = kbuf[t * 16 + nlo];
                const int du = (h * 32 + koff) >> 1;
                uint4 c0 = *(const uint4*)(kr + du);
                uint4 c1 = *(const uint4*)(kr + du + 8);
                v16bf b = mk_v16bf(c0, c1);
                c = __builtin_amdgcn_wmma_f32_16x16x32_bf16(false, qa[h], false, b,
                                                            (short)0, c, false, false);
            }
            sc[t] = c;
        }
        __syncthreads();   // kbuf fully consumed; pbuf (aliased) may be written

        // ---- online softmax (C layout: VGPR g -> row g+8*(lane>=16), col = nlo)
#pragma unroll
        for (int g = 0; g < 8; ++g) {
            float tm = -3.0e38f;
#pragma unroll
            for (int t = 0; t < 8; ++t) tm = fmaxf(tm, sc[t][g]);
            tm *= SCALE;
#pragma unroll
            for (int m = 1; m < 16; m <<= 1) tm = fmaxf(tm, __shfl_xor(tm, m, 32));
            const float mnew  = fmaxf(mrun[g], tm);
            const float alpha = __expf(mrun[g] - mnew);
            mrun[g] = mnew;
            float ps = 0.f;
#pragma unroll
            for (int t = 0; t < 8; ++t) {
                float p = __expf(sc[t][g] * SCALE - mnew);
                sc[t][g] = p;
                ps += p;
            }
#pragma unroll
            for (int m = 1; m < 16; m <<= 1) ps += __shfl_xor(ps, m, 32);
            lrun[g] = lrun[g] * alpha + ps;
#pragma unroll
            for (int t = 0; t < 4; ++t) oacc[t][g] *= alpha;
        }

        // ---- probs: C layout -> LDS -> A layout (per-wave buffer, no barrier)
        {
            const int rowoff = (lane < 16) ? 0 : 8;
#pragma unroll
            for (int g = 0; g < 8; ++g)
#pragma unroll
                for (int t = 0; t < 8; ++t)
                    pbuf[wave][rowoff + g][t * 16 + nlo] = (__bf16)sc[t][g];
        }
        v16bf pa[4];
#pragma unroll
        for (int kb = 0; kb < 4; ++kb) {
            const __bf16* prow = pbuf[wave][nlo];
            uint4 c0 = *(const uint4*)(prow + kb * 32 + koff);
            uint4 c1 = *(const uint4*)(prow + kb * 32 + koff + 16);
            pa[kb] = mk_v16bf(c0, c1);
        }

        // ---- O += P V : 4 dim tiles x 4 key blocks
#pragma unroll
        for (int nd = 0; nd < 4; ++nd) {
#pragma unroll
            for (int kb = 0; kb < 4; ++kb) {
                const __bf16* vr = vbuf[nd * 16 + nlo];
                uint4 c0 = *(const uint4*)(vr + kb * 32 + koff);
                uint4 c1 = *(const uint4*)(vr + kb * 32 + koff + 16);
                v16bf vb = mk_v16bf(c0, c1);
                oacc[nd] = __builtin_amdgcn_wmma_f32_16x16x32_bf16(false, pa[kb], false, vb,
                                                                   (short)0, oacc[nd], false, false);
            }
        }
    }

    // ---- epilogue: normalize, weight by 1/3, scatter back (dilated)
    const int rowoff = (lane < 16) ? 0 : 8;
#pragma unroll
    for (int g = 0; g < 8; ++g) {
        const int  jrow = qtile * 128 + wave * 16 + rowoff + g;
        const long so   = base_s + (long)jrow * r;
        const float inv = 1.f / lrun[g];
        float* orow = out + (so * H + head) * D;
#pragma unroll
        for (int t = 0; t < 4; ++t) {
            const float val  = oacc[t][g] * inv * (1.0f / 3.0f);
            const int   dcol = t * 16 + nlo;
            if (accumulate) orow[dcol] += val;
            else            orow[dcol]  = val;
        }
    }
}

extern "C" void kernel_launch(void* const* d_in, const int* in_sizes, int n_in,
                              void* d_out, int out_size, void* d_ws, size_t ws_size,
                              hipStream_t stream) {
    (void)in_sizes; (void)n_in; (void)out_size; (void)d_ws; (void)ws_size;
    const float* q = (const float*)d_in[0];
    const float* k = (const float*)d_in[1];
    const float* v = (const float*)d_in[2];
    float* out = (float*)d_out;

    // Branch (L=2048, r=1): covers every position -> initializes out = val/3.
    dilated_attn_pass<<<dim3(16, 4, 16), 256, 0, stream>>>(q, k, v, out, 2048, 1, 0);
    // Branch (L=4096, r=2): accumulates into half the positions.
    dilated_attn_pass<<<dim3(16, 2, 16), 256, 0, stream>>>(q, k, v, out, 4096, 2, 1);
    // Branch (L=8192, r=4): accumulates into a quarter of the positions.
    dilated_attn_pass<<<dim3(16, 1, 16), 256, 0, stream>>>(q, k, v, out, 8192, 4, 1);
}
